// PromptedSwinTransformerBlock_85194971283608
// MI455X (gfx1250) — matrix operand order repaired and checked by reading, not compiled, not run on hardware
//
#include <hip/hip_runtime.h>
#include <hip/hip_bf16.h>
#include <math.h>

typedef _Float16 half_t;
typedef __attribute__((ext_vector_type(16))) _Float16 v16h;
typedef __attribute__((ext_vector_type(8)))  float    v8f;

#define WMMA_F32_F16(a, b, c) \
    __builtin_amdgcn_wmma_f32_16x16x32_f16(false, (a), false, (b), (short)0, (c), false, false)

// Build a v16h operand from two 16-byte aligned LDS chunks (matches the wave32
// A/B VGPR layouts from CDNA5 ISA when addressed per-lane).
static inline __device__ v16h ld_v16(const half_t* p0, const half_t* p1) {
    union { v16h v; uint4 u[2]; } x;
    x.u[0] = *(const uint4*)p0;
    x.u[1] = *(const uint4*)p1;
    return x.v;
}

// gfx1250 async copy: 16B per lane, global -> LDS, no VGPR round trip.
// VDST VGPR holds the LDS byte address (low 32 bits of the generic shared ptr).
static inline __device__ void async_ld_b128(const half_t* gsrc, unsigned lds_addr) {
    asm volatile("global_load_async_to_lds_b128 %0, %1, off"
                 :: "v"(lds_addr), "v"(gsrc) : "memory");
}
static inline __device__ void wait_async0() {
    asm volatile("s_wait_asynccnt 0x0" ::: "memory");
}
static inline __device__ unsigned lds_off(const void* p) {
    return (unsigned)(unsigned long long)p;   // LDS aperture: addr[31:0] = LDS offset
}

enum { EPI_QKV = 0, EPI_GELU = 1, EPI_FC2 = 2, EPI_PROJ = 3 };

// ---------------------------------------------------------------------------
// f32 -> f16 convert (weights)
// ---------------------------------------------------------------------------
__global__ void f32_to_f16_kernel(const float* __restrict__ in, half_t* __restrict__ out, int n) {
    int i = blockIdx.x * 256 + threadIdx.x;
    if (i < n) out[i] = (half_t)in[i];
}

// ---------------------------------------------------------------------------
// LayerNorm over C=128, wave per row, 8 rows / block
// ---------------------------------------------------------------------------
__global__ __launch_bounds__(256)
void layernorm_f16_kernel(const float* __restrict__ X, const float* __restrict__ gam,
                          const float* __restrict__ bet, half_t* __restrict__ Y) {
    int row  = blockIdx.x * 8 + (threadIdx.x >> 5);
    int lane = threadIdx.x & 31;
    const float* xr = X + (size_t)row * 128;
    float4 v = *(const float4*)(xr + lane * 4);
    float s = v.x + v.y + v.z + v.w;
#pragma unroll
    for (int m = 16; m >= 1; m >>= 1) s += __shfl_xor(s, m, 32);
    float mu = s * (1.0f / 128.0f);
    float d0 = v.x - mu, d1 = v.y - mu, d2 = v.z - mu, d3 = v.w - mu;
    float q = d0 * d0 + d1 * d1 + d2 * d2 + d3 * d3;
#pragma unroll
    for (int m = 16; m >= 1; m >>= 1) q += __shfl_xor(q, m, 32);
    float is = rsqrtf(q * (1.0f / 128.0f) + 1e-5f);
    float4 g = *(const float4*)(gam + lane * 4);
    float4 b = *(const float4*)(bet + lane * 4);
    half_t* yr = Y + (size_t)row * 128;
    yr[lane * 4 + 0] = (half_t)(d0 * is * g.x + b.x);
    yr[lane * 4 + 1] = (half_t)(d1 * is * g.y + b.y);
    yr[lane * 4 + 2] = (half_t)(d2 * is * g.z + b.z);
    yr[lane * 4 + 3] = (half_t)(d3 * is * g.w + b.w);
}

// ---------------------------------------------------------------------------
// WMMA GEMM: C(MxN) = A(MxK,f16) * B(KxN,f16) + epilogue.
// Block tile 64x128, 8 waves, each wave: one 16-row tile x four 16-col tiles
// (4 WMMAs per K-step). A tile staged via async-to-LDS; B staged transposed.
// M % 64 == 0, N % 128 == 0, K % 32 == 0 guaranteed by caller.
// ---------------------------------------------------------------------------
__global__ __launch_bounds__(256)
void gemm_wmma_kernel(const half_t* __restrict__ A, const half_t* __restrict__ Bw,
                      const float* __restrict__ bias, void* __restrict__ Out,
                      float* __restrict__ x2, int M, int N, int K, int epi) {
    __shared__ __align__(16) half_t lds_a[64 * 32];    // A tile, row-major [64][32]
    __shared__ __align__(16) half_t lds_bt[128 * 32];  // B tile transposed [n=128][k=32]
    const int tid  = threadIdx.x;
    const int wv   = tid >> 5, lane = tid & 31;
    const int lr   = lane & 15, hi = lane >> 4;
    const int wm   = wv & 3,  wn = wv >> 2;
    const int bm   = blockIdx.x, bn = blockIdx.y;

    v8f acc[4] = {};

    for (int k0 = 0; k0 < K; k0 += 32) {
        {   // stage A: 64 rows x 32 halves, async 16B per thread
            int row = tid >> 2, c = tid & 3;
            const half_t* gsrc = A + ((size_t)(bm * 64 + row)) * K + k0 + c * 8;
            async_ld_b128(gsrc, lds_off(&lds_a[row * 32 + c * 8]));
        }
        {   // stage B transposed: Bt[n][k], 16 halves per thread
            int kk = tid >> 3, nc = (tid & 7) * 16;
            uint4 u0 = *(const uint4*)(Bw + ((size_t)(k0 + kk)) * N + bn * 128 + nc);
            uint4 u1 = *(const uint4*)(Bw + ((size_t)(k0 + kk)) * N + bn * 128 + nc + 8);
            half_t tmp[16];
            *(uint4*)tmp       = u0;
            *(uint4*)(tmp + 8) = u1;
#pragma unroll
            for (int j = 0; j < 16; ++j) lds_bt[(nc + j) * 32 + kk] = tmp[j];
        }
        wait_async0();
        __syncthreads();
        const half_t* ar = &lds_a[(wm * 16 + lr) * 32];
        v16h a = ld_v16(ar + hi * 8, ar + 16 + hi * 8);
#pragma unroll
        for (int j = 0; j < 4; ++j) {
            const half_t* br = &lds_bt[(wn * 64 + j * 16 + lr) * 32 + hi * 8];
            v16h b = ld_v16(br, br + 16);
            acc[j] = WMMA_F32_F16(a, b, acc[j]);
        }
        __syncthreads();
    }

    const int rowbase = bm * 64 + wm * 16;
    const int colbase = bn * 128 + wn * 64;
#pragma unroll
    for (int j = 0; j < 4; ++j) {
#pragma unroll
        for (int rr = 0; rr < 8; ++rr) {
            int row = rowbase + rr + hi * 8;
            int col = colbase + j * 16 + lr;
            float val = acc[j][rr] + bias[col];
            if (epi == EPI_QKV) {
                if (col < 128) val *= 0.17677669529663687f;  // HD^-0.5
                ((half_t*)Out)[(size_t)row * N + col] = (half_t)val;
            } else if (epi == EPI_GELU) {
                float g = 0.5f * val * (1.0f + erff(val * 0.7071067811865475f));
                ((half_t*)Out)[(size_t)row * N + col] = (half_t)g;
            } else if (epi == EPI_FC2) {
                ((float*)Out)[(size_t)row * N + col] = val + x2[(size_t)row * N + col];
            } else {  // EPI_PROJ: scatter into residual stream x2
                int win = row / 54, tok = row - win * 54;
                if (tok < 5) {
                    // prompt rows were ordered (w*B + b): b = win % 64; mean over 64 windows
                    size_t dst = ((size_t)(win & 63) * 3141 + tok) * 128 + col;
                    atomicAdd(&x2[dst], val * (1.0f / 64.0f));
                } else {
                    int b = win >> 6, w = win & 63;       // spatial rows: b*NW + w
                    int tt = tok - 5, th = tt / 7, tw = tt - th * 7;
                    int wh = w >> 3, ww = w & 7;
                    int gh = wh * 7 + th + 3; if (gh >= 56) gh -= 56;  // inverse roll
                    int gw = ww * 7 + tw + 3; if (gw >= 56) gw -= 56;
                    size_t dst = ((size_t)b * 3141 + 5 + gh * 56 + gw) * 128 + col;
                    atomicAdd(&x2[dst], val);
                }
            }
        }
    }
}

// ---------------------------------------------------------------------------
// Windowed attention: 1 block = 1 window (4096 windows), 512 threads = 16 waves
// = 4 heads x 4 row tiles of 16 (N=54 padded to 64). HD=32.
// ---------------------------------------------------------------------------
__global__ __launch_bounds__(512)
void attn_window_kernel(const half_t* __restrict__ qkv, const float* __restrict__ rpb,
                        half_t* __restrict__ outp) {
    __shared__ __align__(16) char smem[48 * 1024];
    half_t* qs  = (half_t*)smem;               // [4][64][32]
    half_t* ks  = (half_t*)(smem + 16 * 1024); // [4][64][32]
    half_t* vts = (half_t*)(smem + 32 * 1024); // [4][32][64]  V transposed
    half_t* ps  = (half_t*)smem;               // [4][64][64]  probs, overlays qs+ks

    const int r   = blockIdx.x;
    const int b_s = r >> 6, w = r & 63;   // spatial rows: b*NW + w
    const int b_p = r & 63;               // prompt rows were (w*B + b) -> b = r % 64
    const int wh  = w >> 3, ww = w & 7;
    const int tid = threadIdx.x;

    // Stage q/k/v for all heads: 54 tokens x 16 chunks of 8 halves.
    // q,k are straight copies -> async-to-LDS; v needs a transpose -> sync.
    for (int idx = tid; idx < 54 * 16; idx += 512) {
        int t = idx >> 4, cc = idx & 15;
        int hd = cc >> 2, d0 = (cc & 3) * 8;
        size_t grow;
        if (t < 5) {
            grow = (size_t)b_p * 3141 + t;
        } else {
            int tt = t - 5, th = tt / 7, tw = tt - th * 7;
            int gh = wh * 7 + th + 3; if (gh >= 56) gh -= 56;  // shifted gather
            int gw = ww * 7 + tw + 3; if (gw >= 56) gw -= 56;
            grow = (size_t)b_s * 3141 + 5 + gh * 56 + gw;
        }
        const half_t* base = qkv + grow * 384 + hd * 32 + d0;
        async_ld_b128(base,       lds_off(&qs[(hd * 64 + t) * 32 + d0]));
        async_ld_b128(base + 128, lds_off(&ks[(hd * 64 + t) * 32 + d0]));
        uint4 uv = *(const uint4*)(base + 256);
        half_t tv[8]; *(uint4*)tv = uv;
#pragma unroll
        for (int jj = 0; jj < 8; ++jj) vts[(hd * 32 + d0 + jj) * 64 + t] = tv[jj];
    }
    // Zero pad rows/cols 54..63
    for (int idx = tid; idx < 4 * 10 * 32; idx += 512) {
        int hd = idx / 320, rem = idx - hd * 320;
        int t = 54 + rem / 32, d = rem & 31;
        qs[(hd * 64 + t) * 32 + d]  = (half_t)0.f;
        ks[(hd * 64 + t) * 32 + d]  = (half_t)0.f;
        vts[(hd * 32 + d) * 64 + t] = (half_t)0.f;
    }
    wait_async0();
    __syncthreads();

    const int wv = tid >> 5, lane = tid & 31;
    const int lr = lane & 15, hi = lane >> 4;
    const int h  = wv >> 2,  i  = wv & 3;
    const half_t* qh = qs  + h * 64 * 32;
    const half_t* kh = ks  + h * 64 * 32;
    const half_t* vh = vts + h * 32 * 64;
    half_t*       ph = ps  + h * 64 * 64;

    // S = q @ k^T  (K-dim 32, one WMMA per 16x16 tile)
    v8f s[4];
    {
        const half_t* arow = qh + (i * 16 + lr) * 32;
        v16h a = ld_v16(arow + hi * 8, arow + 16 + hi * 8);
#pragma unroll
        for (int j = 0; j < 4; ++j) {
            const half_t* brow = kh + (j * 16 + lr) * 32 + hi * 8;
            v16h b = ld_v16(brow, brow + 16);
            v8f z = {};
            s[j] = WMMA_F32_F16(a, b, z);
        }
    }
    // relative-position bias + shift mask (computed arithmetically)
#pragma unroll
    for (int j = 0; j < 4; ++j) {
#pragma unroll
        for (int rr = 0; rr < 8; ++rr) {
            int tm = i * 16 + rr + hi * 8;
            int tn = j * 16 + lr;
            float bia;
            if (tn >= 54)                 bia = -1e30f;   // padded columns -> prob 0
            else if (tm >= 54)            bia = 0.0f;     // padded rows, discarded
            else if (tm < 5 || tn < 5)    bia = 0.0f;     // prompt pad region
            else {
                int t1 = tm - 5, t2 = tn - 5;
                int th1 = t1 / 7, tw1 = t1 - th1 * 7;
                int th2 = t2 / 7, tw2 = t2 - th2 * 7;
                float rv = rpb[((th1 - th2 + 6) * 13 + (tw1 - tw2 + 6)) * 4 + h];
                int gh1 = wh * 7 + th1, gw1 = ww * 7 + tw1;
                int gh2 = wh * 7 + th2, gw2 = ww * 7 + tw2;
                int r1 = (gh1 < 49 ? 0 : (gh1 < 53 ? 1 : 2)) * 3 + (gw1 < 49 ? 0 : (gw1 < 53 ? 1 : 2));
                int r2 = (gh2 < 49 ? 0 : (gh2 < 53 ? 1 : 2)) * 3 + (gw2 < 49 ? 0 : (gw2 < 53 ? 1 : 2));
                bia = rv + ((r1 != r2) ? -100.0f : 0.0f);
            }
            s[j][rr] += bia;
        }
    }
    // Row softmax directly on the C-layout registers (16-lane segments = one row)
    half_t pr[4][8];
#pragma unroll
    for (int rr = 0; rr < 8; ++rr) {
        float mx = fmaxf(fmaxf(s[0][rr], s[1][rr]), fmaxf(s[2][rr], s[3][rr]));
#pragma unroll
        for (int m = 8; m >= 1; m >>= 1) mx = fmaxf(mx, __shfl_xor(mx, m, 16));
        float e0 = __expf(s[0][rr] - mx), e1 = __expf(s[1][rr] - mx);
        float e2 = __expf(s[2][rr] - mx), e3 = __expf(s[3][rr] - mx);
        float sm = e0 + e1 + e2 + e3;
#pragma unroll
        for (int m = 8; m >= 1; m >>= 1) sm += __shfl_xor(sm, m, 16);
        float inv = 1.0f / sm;
        pr[0][rr] = (half_t)(e0 * inv); pr[1][rr] = (half_t)(e1 * inv);
        pr[2][rr] = (half_t)(e2 * inv); pr[3][rr] = (half_t)(e3 * inv);
    }
    __syncthreads();  // all waves done reading qs/ks before overlay write
#pragma unroll
    for (int j = 0; j < 4; ++j)
#pragma unroll
        for (int rr = 0; rr < 8; ++rr) {
            int tm = i * 16 + rr + hi * 8, tn = j * 16 + lr;
            ph[tm * 64 + tn] = pr[j][rr];
        }
    __syncthreads();

    // O = P @ V   (K-dim 64 -> 2 WMMAs per 16x16 output tile, 2 col tiles)
#pragma unroll
    for (int jd = 0; jd < 2; ++jd) {
        v8f o = {};
#pragma unroll
        for (int kk = 0; kk < 2; ++kk) {
            const half_t* arow = ph + (i * 16 + lr) * 64 + kk * 32;
            v16h a = ld_v16(arow + hi * 8, arow + 16 + hi * 8);
            const half_t* brow = vh + (jd * 16 + lr) * 64 + kk * 32 + hi * 8;
            v16h b = ld_v16(brow, brow + 16);
            o = WMMA_F32_F16(a, b, o);
        }
#pragma unroll
        for (int rr = 0; rr < 8; ++rr) {
            int tm = i * 16 + rr + hi * 8;
            if (tm < 54) {
                int col = h * 32 + jd * 16 + lr;
                outp[((size_t)r * 54 + tm) * 128 + col] = (half_t)o[rr];
            }
        }
    }
}

// ---------------------------------------------------------------------------
extern "C" void kernel_launch(void* const* d_in, const int* in_sizes, int n_in,
                              void* d_out, int out_size, void* d_ws, size_t ws_size,
                              hipStream_t stream) {
    const float* x      = (const float*)d_in[0];
    const float* n1g    = (const float*)d_in[1];
    const float* n1b    = (const float*)d_in[2];
    const float* qkv_w  = (const float*)d_in[3];
    const float* qkv_b  = (const float*)d_in[4];
    const float* proj_w = (const float*)d_in[5];
    const float* proj_b = (const float*)d_in[6];
    const float* rpb    = (const float*)d_in[7];
    const float* n2g    = (const float*)d_in[8];
    const float* n2b    = (const float*)d_in[9];
    const float* fc1_w  = (const float*)d_in[10];
    const float* fc1_b  = (const float*)d_in[11];
    const float* fc2_w  = (const float*)d_in[12];
    const float* fc2_b  = (const float*)d_in[13];

    char* ws = (char*)d_ws;
    // Arena (bytes), all offsets 256-aligned:
    half_t* xn_h    = (half_t*)(ws);                         // 201024*128*2 = 51,462,144 (reused for LN2 out)
    half_t* qkv_h   = (half_t*)(ws + 51462144);              // 201024*384*2 = 154,386,432
    half_t* attn_h  = (half_t*)(ws + 205848576);             // 221184*128*2 =  56,623,104
    half_t* hmid_h  = (half_t*)(ws + 51462144);              // reuse qkv+attn region: 201024*512*2 = 205,848,576
    float*  x2      = (float*)(ws + 262471680);              // 201024*128*4 = 102,924,288
    half_t* qkvw_h  = (half_t*)(ws + 365395968);             // 49152*2
    half_t* projw_h = (half_t*)(ws + 365395968 + 98304);     // 16384*2
    half_t* fc1w_h  = (half_t*)(ws + 365395968 + 131072);    // 65536*2
    half_t* fc2w_h  = (half_t*)(ws + 365395968 + 262144);    // 65536*2

    // residual stream starts as the shortcut
    hipMemcpyAsync(x2, x, 102924288, hipMemcpyDeviceToDevice, stream);

    f32_to_f16_kernel<<<(49152 + 255) / 256, 256, 0, stream>>>(qkv_w, qkvw_h, 49152);
    f32_to_f16_kernel<<<(16384 + 255) / 256, 256, 0, stream>>>(proj_w, projw_h, 16384);
    f32_to_f16_kernel<<<(65536 + 255) / 256, 256, 0, stream>>>(fc1_w, fc1w_h, 65536);
    f32_to_f16_kernel<<<(65536 + 255) / 256, 256, 0, stream>>>(fc2_w, fc2w_h, 65536);

    // LN1 -> xn (f16)
    layernorm_f16_kernel<<<25128, 256, 0, stream>>>(x, n1g, n1b, xn_h);
    // QKV: [201024,128] @ [128,384]
    gemm_wmma_kernel<<<dim3(3141, 3), 256, 0, stream>>>(xn_h, qkvw_h, qkv_b, qkv_h,
                                                        nullptr, 201024, 384, 128, EPI_QKV);
    // windowed attention -> attn_h [4096*54,128]
    attn_window_kernel<<<4096, 512, 0, stream>>>(qkv_h, rpb, attn_h);
    // proj: [221184,128] @ [128,128], scatter-add into x2
    gemm_wmma_kernel<<<dim3(3456, 1), 256, 0, stream>>>(attn_h, projw_h, proj_b, nullptr,
                                                        x2, 221184, 128, 128, EPI_PROJ);
    // LN2 -> (reuse xn_h)
    layernorm_f16_kernel<<<25128, 256, 0, stream>>>(x2, n2g, n2b, xn_h);
    // fc1 + GELU: [201024,128] @ [128,512]
    gemm_wmma_kernel<<<dim3(3141, 4), 256, 0, stream>>>(xn_h, fc1w_h, fc1_b, hmid_h,
                                                        nullptr, 201024, 512, 128, EPI_GELU);
    // fc2 + residual: [201024,512] @ [512,128] -> d_out (f32)
    gemm_wmma_kernel<<<dim3(3141, 1), 256, 0, stream>>>(hmid_h, fc2w_h, fc2_b, (float*)d_out,
                                                        x2, 201024, 128, 512, EPI_FC2);
}
